// KHSTDenoiser_37409165148279
// MI455X (gfx1250) — matrix operand order repaired and checked by reading, not compiled, not run on hardware
//
#include <hip/hip_runtime.h>
#include <hip/hip_bf16.h>
#include <math.h>

// ---------------------------------------------------------------------------
// KHSTDenoiser for MI455X (gfx1250): 4-layer transformer block stack.
// GEMMs run on v_wmma_f32_16x16x32_f16 (f16 operands, f32 accumulate).
// Wave tile 32M x 64N (8 accumulators); LDS double-buffered so the next
// K-slice's global loads overlap the current 8-WMMA chain.
// ---------------------------------------------------------------------------

#define DEPTH   4
#define HEADS   8
#define DIM     512
#define HIDDEN  1024
#define SEQ     17
#define BATCH   2048
#define MROWS   (BATCH * SEQ)          // 34816, divisible by 64
#define K_HOPS  4

typedef __attribute__((ext_vector_type(16))) _Float16 v16h;
typedef __attribute__((ext_vector_type(8)))  float    v8f;

union V16HU { uint4 u4[2]; v16h v; };

// ---------------------------------------------------------------------------
// fp32 -> fp16 conversion (weights, one-time)
// ---------------------------------------------------------------------------
__global__ void cvt_f32_to_f16_kernel(const float* __restrict__ src,
                                      _Float16* __restrict__ dst, int n) {
    int i = blockIdx.x * blockDim.x + threadIdx.x;
    int stride = gridDim.x * blockDim.x;
    for (; i < n; i += stride) dst[i] = (_Float16)src[i];
}

// ---------------------------------------------------------------------------
// hop-bias precompute: bias[h,i,j] = rel_alpha[h] * sum_k softmax(logits[h])[k] * H[k,i,j]
// ---------------------------------------------------------------------------
__global__ void hop_bias_kernel(const float* __restrict__ Hstack,      // [K,17,17]
                                const float* __restrict__ hop_logits,  // [H,K]
                                const float* __restrict__ rel_alpha,   // [H]
                                float* __restrict__ bias) {            // [H,17,17]
    int idx = blockIdx.x * blockDim.x + threadIdx.x;
    const int NN = SEQ * SEQ;
    if (idx >= HEADS * NN) return;
    int h  = idx / NN;
    int ij = idx % NN;
    float l[K_HOPS], mx = -1e30f;
    #pragma unroll
    for (int k = 0; k < K_HOPS; ++k) { l[k] = hop_logits[h * K_HOPS + k]; mx = fmaxf(mx, l[k]); }
    float s = 0.f;
    #pragma unroll
    for (int k = 0; k < K_HOPS; ++k) { l[k] = __expf(l[k] - mx); s += l[k]; }
    float acc = 0.f;
    #pragma unroll
    for (int k = 0; k < K_HOPS; ++k) acc += (l[k] / s) * Hstack[k * NN + ij];
    bias[idx] = rel_alpha[h] * acc;
}

// ---------------------------------------------------------------------------
// LayerNorm row kernel: fp32 [M,512] -> fp16 [M,512].  One block (128 thr) / row.
// ---------------------------------------------------------------------------
__global__ void ln_f16_kernel(const float* __restrict__ x,
                              const float* __restrict__ g,
                              const float* __restrict__ b,
                              _Float16* __restrict__ out) {
    const int row  = blockIdx.x;
    const int tid  = threadIdx.x;          // 0..127, 4 floats each
    const int lane = tid & 31;
    const int wid  = tid >> 5;

    float4 v = ((const float4*)(x + (size_t)row * DIM))[tid];
    float s  = v.x + v.y + v.z + v.w;
    float ss = v.x * v.x + v.y * v.y + v.z * v.z + v.w * v.w;
    #pragma unroll
    for (int off = 16; off > 0; off >>= 1) {
        s  += __shfl_xor(s,  off, 32);
        ss += __shfl_xor(ss, off, 32);
    }
    __shared__ float sA[4], sB[4];
    if (lane == 0) { sA[wid] = s; sB[wid] = ss; }
    __syncthreads();
    float tot  = sA[0] + sA[1] + sA[2] + sA[3];
    float tot2 = sB[0] + sB[1] + sB[2] + sB[3];
    float mean = tot * (1.0f / DIM);
    float var  = tot2 * (1.0f / DIM) - mean * mean;
    float rstd = rsqrtf(var + 1e-5f);

    float4 gv = ((const float4*)g)[tid];
    float4 bv = ((const float4*)b)[tid];
    _Float16* o = out + (size_t)row * DIM + tid * 4;
    o[0] = (_Float16)((v.x - mean) * rstd * gv.x + bv.x);
    o[1] = (_Float16)((v.y - mean) * rstd * gv.y + bv.y);
    o[2] = (_Float16)((v.z - mean) * rstd * gv.z + bv.z);
    o[3] = (_Float16)((v.w - mean) * rstd * gv.w + bv.w);
}

// ---------------------------------------------------------------------------
// WMMA GEMM:  D[M,Nw] = A[M,K] (f16) @ W[Nw,K]^T (f16) + bias
//   mode 0: store f16
//   mode 1: store f16 after exact GELU
//   mode 2: fp32 in-place residual:  outF[i] += D[i]
// Block: 256 thr (8 waves). Block tile 64M x 256N; wave = 32M x 64N
// (2 A-frags x 4 B-frags -> 8 WMMAs per K-step). LDS double buffered.
// ---------------------------------------------------------------------------
__global__ __launch_bounds__(256)
void gemm_wmma_kernel(const _Float16* __restrict__ A,
                      const _Float16* __restrict__ W,
                      const float* __restrict__ bias,
                      _Float16* __restrict__ outH,
                      float* __restrict__ outF,
                      int M, int Nw, int K, int mode) {
    __shared__ __attribute__((aligned(16))) _Float16 As[2][64 * 32];    // 2 x 4 KB
    __shared__ __attribute__((aligned(16))) _Float16 Bs[2][256 * 32];   // 2 x 16 KB

    const int tid  = threadIdx.x;
    const int lane = tid & 31;
    const int wid  = tid >> 5;
    const int mt   = wid & 1;    // wave's 32-row half within 64
    const int ng   = wid >> 1;   // wave's 64-col group within 256

    const int m0 = blockIdx.y * 64;
    const int n0 = blockIdx.x * 256;

    v8f acc[2][4];
    #pragma unroll
    for (int mi = 0; mi < 2; ++mi)
        #pragma unroll
        for (int j = 0; j < 4; ++j) acc[mi][j] = (v8f){};

    // staging geometry: 1 uint4 (8 halves) per thread for A, 4 for B
    const int srow = tid >> 2;           // 0..63
    const int sseg = tid & 3;            // 0..3  (halves [seg*8, seg*8+8))
    const _Float16* aG = A + (size_t)(m0 + srow) * K + sseg * 8;
    const _Float16* bG = W + (size_t)(n0 + srow) * K + sseg * 8;
    const size_t bRep = (size_t)64 * K;  // 64 W-rows per rep
    const int sOff = srow * 32 + sseg * 8;

    // operand lane mappings (ISA 16-bit 16x32 A / 32x16 B layouts)
    const int akb = (lane >> 4) * 8;     // 0 or 8 halves
    const int bkb = (lane >> 4) * 8;     // 0 or 8 halves

    const int nk = K >> 5;               // K / 32 steps

    // ---- prologue: stage K-slice 0 into buffer 0 ----
    {
        uint4 ta = *(const uint4*)aG;
        uint4 tb0 = *(const uint4*)(bG);
        uint4 tb1 = *(const uint4*)(bG + bRep);
        uint4 tb2 = *(const uint4*)(bG + 2 * bRep);
        uint4 tb3 = *(const uint4*)(bG + 3 * bRep);
        *(uint4*)(As[0] + sOff) = ta;
        *(uint4*)(Bs[0] + sOff) = tb0;
        *(uint4*)(Bs[0] + 64 * 32 + sOff) = tb1;
        *(uint4*)(Bs[0] + 128 * 32 + sOff) = tb2;
        *(uint4*)(Bs[0] + 192 * 32 + sOff) = tb3;
    }
    __syncthreads();

    for (int t = 0; t < nk; ++t) {
        const int buf = t & 1;

        // ---- issue next K-slice's global loads before this step's WMMAs ----
        uint4 na, nb0, nb1, nb2, nb3;
        const bool more = (t + 1) < nk;
        if (more) {
            const int ko = (t + 1) * 32;
            na  = *(const uint4*)(aG + ko);
            nb0 = *(const uint4*)(bG + ko);
            nb1 = *(const uint4*)(bG + bRep + ko);
            nb2 = *(const uint4*)(bG + 2 * bRep + ko);
            nb3 = *(const uint4*)(bG + 3 * bRep + ko);
            if (t + 2 < nk) {
                // hint the K-slice after that into cache (global_prefetch_b8)
                __builtin_prefetch((const void*)(aG + (t + 2) * 32), 0, 1);
                __builtin_prefetch((const void*)(bG + (t + 2) * 32), 0, 1);
            }
        }

        // ---- build the wave's two A fragments (reused by 4 column tiles) ----
        V16HU af[2];
        #pragma unroll
        for (int mi = 0; mi < 2; ++mi) {
            const int ar = mt * 32 + mi * 16 + (lane & 15);
            const uint4* pa = (const uint4*)(As[buf] + ar * 32);
            af[mi].u4[0] = pa[akb >> 3];        // halves [akb, akb+8)
            af[mi].u4[1] = pa[2 + (akb >> 3)];  // halves [16+akb, 16+akb+8)
        }
        // ---- 8 WMMAs: 4 column tiles x 2 row tiles ----
        #pragma unroll
        for (int j = 0; j < 4; ++j) {
            V16HU bf;
            const int cb = ng * 64 + j * 16 + (lane & 15);
            const uint4* pb = (const uint4*)(Bs[buf] + cb * 32 + bkb);
            bf.u4[0] = pb[0];
            bf.u4[1] = pb[2];
            acc[0][j] = __builtin_amdgcn_wmma_f32_16x16x32_f16(
                false, af[0].v, false, bf.v, (short)0, acc[0][j], false, false);
            acc[1][j] = __builtin_amdgcn_wmma_f32_16x16x32_f16(
                false, af[1].v, false, bf.v, (short)0, acc[1][j], false, false);
        }
        __syncthreads();     // everyone done reading buf (and buf^1 from t-1)

        if (more) {
            const int nbuf = buf ^ 1;
            *(uint4*)(As[nbuf] + sOff) = na;
            *(uint4*)(Bs[nbuf] + sOff) = nb0;
            *(uint4*)(Bs[nbuf] + 64 * 32 + sOff) = nb1;
            *(uint4*)(Bs[nbuf] + 128 * 32 + sOff) = nb2;
            *(uint4*)(Bs[nbuf] + 192 * 32 + sOff) = nb3;
            __syncthreads(); // stores visible before next step's fragment reads
        }
    }

    // ---- epilogue: C/D layout -> row = (lane>>4)*8 + i, col = lane&15 ----
    #pragma unroll
    for (int mi = 0; mi < 2; ++mi) {
        const int rbase = m0 + mt * 32 + mi * 16 + (lane >> 4) * 8;
        #pragma unroll
        for (int j = 0; j < 4; ++j) {
            const int col = n0 + ng * 64 + j * 16 + (lane & 15);
            const float bv = bias[col];
            #pragma unroll
            for (int i = 0; i < 8; ++i) {
                const size_t idx = (size_t)(rbase + i) * Nw + col;
                float v = acc[mi][j][i] + bv;
                if (mode == 2) {
                    outF[idx] = outF[idx] + v;              // fp32 residual add
                } else {
                    if (mode == 1)                          // exact GELU
                        v = 0.5f * v * (1.0f + erff(v * 0.70710678118654752f));
                    outH[idx] = (_Float16)v;
                }
            }
        }
    }
}

// ---------------------------------------------------------------------------
// Attention: per batch element b (block, 256 thr), one wave per head.
// qkv f16 [M, 1536] (q|k|v each 512 wide, head h at cols h*64), bias [H,17,17].
// Writes f16 [M, 512] already merged across heads.
// ---------------------------------------------------------------------------
__global__ __launch_bounds__(256)
void attn_kernel(const _Float16* __restrict__ qkv,
                 const float* __restrict__ bias,
                 _Float16* __restrict__ out) {
    __shared__ __attribute__((aligned(16))) _Float16 s[SEQ * 3 * DIM];  // 52224 B
    const int b    = blockIdx.x;
    const int tid  = threadIdx.x;
    const int lane = tid & 31;
    const int h    = tid >> 5;

    // stage this batch element's full QKV (17 x 1536 halves)
    {
        const uint4* src = (const uint4*)(qkv + (size_t)b * SEQ * 3 * DIM);
        uint4* dst = (uint4*)s;
        const int nvec = SEQ * 3 * DIM / 8;     // 3264
        for (int i = tid; i < nvec; i += 256) dst[i] = src[i];
    }
    __syncthreads();

    if (lane < SEQ) {
        const int n = lane;
        const float scale = 0.125f;             // 64^-0.5
        const _Float16* qp = s + n * (3 * DIM) + h * 64;

        float logit[SEQ];
        float mx = -1e30f;
        #pragma unroll 1
        for (int m = 0; m < SEQ; ++m) {
            const _Float16* kp = s + m * (3 * DIM) + DIM + h * 64;
            float d = 0.f;
            #pragma unroll
            for (int dd = 0; dd < 64; ++dd) d += (float)qp[dd] * (float)kp[dd];
            d = d * scale + bias[h * (SEQ * SEQ) + n * SEQ + m];
            logit[m] = d;
            mx = fmaxf(mx, d);
        }
        float denom = 0.f;
        #pragma unroll
        for (int m = 0; m < SEQ; ++m) { logit[m] = __expf(logit[m] - mx); denom += logit[m]; }
        const float inv = 1.0f / denom;

        _Float16* op = out + ((size_t)b * SEQ + n) * DIM + h * 64;
        #pragma unroll 1
        for (int dd = 0; dd < 64; ++dd) {
            float o = 0.f;
            #pragma unroll
            for (int m = 0; m < SEQ; ++m)
                o += logit[m] * (float)s[m * (3 * DIM) + 2 * DIM + h * 64 + dd];
            op[dd] = (_Float16)(o * inv);
        }
    }
}

// ---------------------------------------------------------------------------
// Host orchestration
// ---------------------------------------------------------------------------
extern "C" void kernel_launch(void* const* d_in, const int* in_sizes, int n_in,
                              void* d_out, int out_size, void* d_ws, size_t ws_size,
                              hipStream_t stream) {
    const float* x_in       = (const float*)d_in[0];
    const float* Hstack     = (const float*)d_in[1];
    const float* hop_logits = (const float*)d_in[2];
    const float* rel_alpha  = (const float*)d_in[3];
    const float* qkv_w      = (const float*)d_in[4];
    const float* qkv_b      = (const float*)d_in[5];
    const float* proj_w     = (const float*)d_in[6];
    const float* proj_b     = (const float*)d_in[7];
    const float* ln1_g      = (const float*)d_in[8];
    const float* ln1_b      = (const float*)d_in[9];
    const float* ln2_g      = (const float*)d_in[10];
    const float* ln2_b      = (const float*)d_in[11];
    const float* fc1_w      = (const float*)d_in[12];
    const float* fc1_b      = (const float*)d_in[13];
    const float* fc2_w      = (const float*)d_in[14];
    const float* fc2_b      = (const float*)d_in[15];
    (void)in_sizes; (void)n_in; (void)out_size; (void)ws_size;

    float* x = (float*)d_out;   // x lives in d_out, updated in place per layer

    // ---- workspace carve-up (fp16 regions then fp32 bias) ----
    _Float16* ws16 = (_Float16*)d_ws;
    const size_t nqkvw = (size_t)DEPTH * 3 * DIM * DIM;       // 3.15M
    const size_t nprjw = (size_t)DEPTH * DIM * DIM;           // 1.05M
    const size_t nfc1w = (size_t)DEPTH * HIDDEN * DIM;        // 2.10M
    const size_t nfc2w = (size_t)DEPTH * DIM * HIDDEN;        // 2.10M
    const size_t nact  = (size_t)MROWS * DIM;                 // 17.8M
    const size_t nqkva = (size_t)MROWS * 3 * DIM;             // 53.5M (reused for h2)

    _Float16* wqkv  = ws16;                 ws16 += nqkvw;
    _Float16* wproj = ws16;                 ws16 += nprjw;
    _Float16* wfc1  = ws16;                 ws16 += nfc1w;
    _Float16* wfc2  = ws16;                 ws16 += nfc2w;
    _Float16* act   = ws16;                 ws16 += nact;
    _Float16* qkva  = ws16;                 ws16 += nqkva;
    float*    biasB = (float*)(((uintptr_t)ws16 + 255) & ~(uintptr_t)255);

    // ---- x <- input ----
    hipMemcpyAsync(x, x_in, (size_t)MROWS * DIM * sizeof(float),
                   hipMemcpyDeviceToDevice, stream);

    // ---- weights -> fp16 (one time, deterministic each call) ----
    cvt_f32_to_f16_kernel<<<2048, 256, 0, stream>>>(qkv_w, wqkv, (int)nqkvw);
    cvt_f32_to_f16_kernel<<<2048, 256, 0, stream>>>(proj_w, wproj, (int)nprjw);
    cvt_f32_to_f16_kernel<<<2048, 256, 0, stream>>>(fc1_w, wfc1, (int)nfc1w);
    cvt_f32_to_f16_kernel<<<2048, 256, 0, stream>>>(fc2_w, wfc2, (int)nfc2w);

    // ---- hop bias ----
    hop_bias_kernel<<<(HEADS * SEQ * SEQ + 255) / 256, 256, 0, stream>>>(
        Hstack, hop_logits, rel_alpha, biasB);

    const dim3 blk(256);
    for (int d = 0; d < DEPTH; ++d) {
        const _Float16* wq = wqkv  + (size_t)d * 3 * DIM * DIM;
        const _Float16* wp = wproj + (size_t)d * DIM * DIM;
        const _Float16* w1 = wfc1  + (size_t)d * HIDDEN * DIM;
        const _Float16* w2 = wfc2  + (size_t)d * DIM * HIDDEN;

        // LN1 -> f16
        ln_f16_kernel<<<MROWS, 128, 0, stream>>>(x, ln1_g + d * DIM, ln1_b + d * DIM, act);
        // QKV GEMM (f16 out)
        gemm_wmma_kernel<<<dim3(3 * DIM / 256, MROWS / 64), blk, 0, stream>>>(
            act, wq, qkv_b + (size_t)d * 3 * DIM, qkva, nullptr,
            MROWS, 3 * DIM, DIM, 0);
        // attention (f16 out, heads merged)
        attn_kernel<<<BATCH, 256, 0, stream>>>(qkva, biasB, act);
        // proj GEMM + fp32 residual into x
        gemm_wmma_kernel<<<dim3(DIM / 256, MROWS / 64), blk, 0, stream>>>(
            act, wp, proj_b + (size_t)d * DIM, nullptr, x,
            MROWS, DIM, DIM, 2);
        // LN2 -> f16
        ln_f16_kernel<<<MROWS, 128, 0, stream>>>(x, ln2_g + d * DIM, ln2_b + d * DIM, act);
        // FC1 GEMM + exact GELU (f16 out into qkva region)
        gemm_wmma_kernel<<<dim3(HIDDEN / 256, MROWS / 64), blk, 0, stream>>>(
            act, w1, fc1_b + (size_t)d * HIDDEN, qkva, nullptr,
            MROWS, HIDDEN, DIM, 1);
        // FC2 GEMM + fp32 residual into x
        gemm_wmma_kernel<<<dim3(DIM / 256, MROWS / 64), blk, 0, stream>>>(
            qkva, w2, fc2_b + (size_t)d * DIM, nullptr, x,
            MROWS, DIM, HIDDEN, 2);
    }
}
